// MultiHeadAttention_31799937859897
// MI455X (gfx1250) — matrix operand order repaired and checked by reading, not compiled, not run on hardware
//
#include <hip/hip_runtime.h>
#include <hip/hip_bf16.h>

// MI455X (gfx1250) implementation of the per-position multi-head "attention"
// reference. Pipeline:
//   K0: one-shot fp32 -> f16 conversion of the four weight matrices (ws)
//   K1: fused Q/K/V projection GEMMs, fp32 A (cvt in regs) x f16 W -> f16 (ws)
//   K2: per-token 16x16 head-gram + softmax + attn*V, fused LayerNorm -> f16 (ws)
//   K3: output projection GEMM, f16 A x f16 Wo -> fp32 d_out
// All GEMMs use v_wmma_f32_16x16x32_f16 with 64x64 tiles per wave32.

#define DM 1024
#define ROWS (8 * 2048)   // 16384 token rows

typedef __attribute__((ext_vector_type(16))) _Float16 v16h;
typedef __attribute__((ext_vector_type(8)))  _Float16 v8h;
typedef __attribute__((ext_vector_type(8)))  float    v8f;

// ---- fragment loaders -------------------------------------------------------

// A fragment (16x32 f16) from fp32 row-major source. X = row base pointer.
// halves[0..7] = X[k0..k0+7], halves[8..15] = X[k0+16..k0+23]
__device__ __forceinline__ v16h a_frag_f32(const float* __restrict__ X, int k0) {
  const float4* p = (const float4*)(X + k0);
  const float4* q = (const float4*)(X + k0 + 16);
  float4 x0 = p[0], x1 = p[1], x2 = q[0], x3 = q[1];
  v16h a;
  a[0]  = (_Float16)x0.x; a[1]  = (_Float16)x0.y; a[2]  = (_Float16)x0.z; a[3]  = (_Float16)x0.w;
  a[4]  = (_Float16)x1.x; a[5]  = (_Float16)x1.y; a[6]  = (_Float16)x1.z; a[7]  = (_Float16)x1.w;
  a[8]  = (_Float16)x2.x; a[9]  = (_Float16)x2.y; a[10] = (_Float16)x2.z; a[11] = (_Float16)x2.w;
  a[12] = (_Float16)x3.x; a[13] = (_Float16)x3.y; a[14] = (_Float16)x3.z; a[15] = (_Float16)x3.w;
  return a;
}

// A fragment from f16 row-major source (two contiguous 16B chunks).
__device__ __forceinline__ v16h a_frag_f16(const _Float16* __restrict__ X, int k0) {
  v8h lo = *(const v8h*)(X + k0);
  v8h hi = *(const v8h*)(X + k0 + 16);
  v16h a;
#pragma unroll
  for (int i = 0; i < 8; ++i) { a[i] = lo[i]; a[i + 8] = hi[i]; }
  return a;
}

// B fragment (32x16 f16) from f16 row-major W (computing X @ W^T):
// halves[0..7] = W[n][kb..kb+7], halves[8..15] = W[n][kb+16..kb+23]
__device__ __forceinline__ v16h b_frag_f16(const _Float16* __restrict__ Wr, int kb) {
  v8h lo = *(const v8h*)(Wr + kb);
  v8h hi = *(const v8h*)(Wr + kb + 16);
  v16h b;
#pragma unroll
  for (int i = 0; i < 8; ++i) { b[i] = lo[i]; b[i + 8] = hi[i]; }
  return b;
}

// ---- K0: one-shot weight conversion fp32 -> f16 -----------------------------
// grid = (DM*DM/(256*8), 4), block = 256; 8 elements per thread
__global__ __launch_bounds__(256) void cvt_w_kernel(
    const float* __restrict__ Wq, const float* __restrict__ Wk,
    const float* __restrict__ Wv, const float* __restrict__ Wo,
    _Float16* __restrict__ Hq, _Float16* __restrict__ Hk,
    _Float16* __restrict__ Hv, _Float16* __restrict__ Ho) {
  const int m = blockIdx.y;
  const float* W = (m == 0) ? Wq : (m == 1) ? Wk : (m == 2) ? Wv : Wo;
  _Float16*    H = (m == 0) ? Hq : (m == 1) ? Hk : (m == 2) ? Hv : Ho;
  const size_t i = ((size_t)blockIdx.x * 256 + threadIdx.x) * 8;
  float4 a = ((const float4*)(W + i))[0];
  float4 b = ((const float4*)(W + i))[1];
  v8h o;
  o[0] = (_Float16)a.x; o[1] = (_Float16)a.y; o[2] = (_Float16)a.z; o[3] = (_Float16)a.w;
  o[4] = (_Float16)b.x; o[5] = (_Float16)b.y; o[6] = (_Float16)b.z; o[7] = (_Float16)b.w;
  *(v8h*)(H + i) = o;
}

// ---- K1: fused QKV projection ----------------------------------------------
// grid = (ROWS/64, DM/64, 3), block = 32 (one wave -> 64x64 tile)
__global__ __launch_bounds__(32) void qkv_gemm_kernel(
    const float* __restrict__ Xq, const float* __restrict__ Xk, const float* __restrict__ Xv,
    const _Float16* __restrict__ Wq, const _Float16* __restrict__ Wk,
    const _Float16* __restrict__ Wv,
    const float* __restrict__ bq, const float* __restrict__ bk, const float* __restrict__ bv,
    _Float16* __restrict__ Oq, _Float16* __restrict__ Ok, _Float16* __restrict__ Ov) {
  const int lane = threadIdx.x;
  const int l16  = lane & 15;
  const int hi   = lane >> 4;
  const int rowbase = blockIdx.x * 64;
  const int colbase = blockIdx.y * 64;
  const int m = blockIdx.z;

  const float*    X = (m == 0) ? Xq : (m == 1) ? Xk : Xv;
  const _Float16* W = (m == 0) ? Wq : (m == 1) ? Wk : Wv;
  const float*    B = (m == 0) ? bq : (m == 1) ? bk : bv;
  _Float16*       O = (m == 0) ? Oq : (m == 1) ? Ok : Ov;

  const float*    Xrow[4];
  const _Float16* Wrow[4];
#pragma unroll
  for (int i = 0; i < 4; ++i) Xrow[i] = X + (size_t)(rowbase + 16 * i + l16) * DM;
#pragma unroll
  for (int j = 0; j < 4; ++j) Wrow[j] = W + (size_t)(colbase + 16 * j + l16) * DM;

  v8f zero = {};
  v8f acc[4][4];
#pragma unroll
  for (int i = 0; i < 4; ++i)
#pragma unroll
    for (int j = 0; j < 4; ++j) acc[i][j] = zero;

  for (int kt = 0; kt < DM; kt += 32) {
    const int ka = kt + hi * 8;
    const int kb = kt + hi * 8;
    v16h a[4];
#pragma unroll
    for (int i = 0; i < 4; ++i) a[i] = a_frag_f32(Xrow[i], ka);
#pragma unroll
    for (int j = 0; j < 4; ++j) {
      v16h b = b_frag_f16(Wrow[j], kb);
#pragma unroll
      for (int i = 0; i < 4; ++i)
        acc[i][j] = __builtin_amdgcn_wmma_f32_16x16x32_f16(
            false, a[i], false, b, (short)0, acc[i][j], false, false);
    }
  }

#pragma unroll
  for (int j = 0; j < 4; ++j) {
    const int col = colbase + 16 * j + l16;
    const float bias = B[col];
#pragma unroll
    for (int i = 0; i < 4; ++i) {
      _Float16* orow = O + (size_t)(rowbase + 16 * i + hi * 8) * DM + col;
#pragma unroll
      for (int r = 0; r < 8; ++r)
        orow[(size_t)r * DM] = (_Float16)(acc[i][j][r] + bias);
    }
  }
}

// ---- K2: per-token head-gram attention + fused LayerNorm --------------------
// grid = ROWS, block = 256. One token row per block.
#define SP 68  // LDS row pitch in halves (64 + 4 pad -> conflict-free stride)
__global__ __launch_bounds__(256) void attn_ln_kernel(
    const _Float16* __restrict__ Q, const _Float16* __restrict__ K,
    const _Float16* __restrict__ V,
    const float* __restrict__ ln_g, const float* __restrict__ ln_b,
    _Float16* __restrict__ Outh) {
  __shared__ _Float16 qs[16 * SP], ks[16 * SP], vs[16 * SP];
  __shared__ float att[256];
  __shared__ float red1[8], red2[8];
  __shared__ float stat[2];

  const int tid = threadIdx.x;
  const size_t base = (size_t)blockIdx.x * DM;

  // cooperative staged load (4 halves / thread) with padded row pitch
  {
    const int hrow = tid >> 4;            // 0..15
    const int d0   = (tid & 15) * 4;      // 0..60
    const int src  = hrow * 64 + d0;
    const int dst  = hrow * SP + d0;
    *(uint2*)(qs + dst) = *(const uint2*)(Q + base + src);
    *(uint2*)(ks + dst) = *(const uint2*)(K + base + src);
    *(uint2*)(vs + dst) = *(const uint2*)(V + base + src);
  }
  __syncthreads();

  // energy[h][g] = sum_d q[h,d]*k[g,d], scaled by 1/sqrt(1024) = 1/32
  const int h = tid >> 4, g = tid & 15;
  {
    const _Float16* qp = qs + h * SP;
    const _Float16* kp = ks + g * SP;
    float e = 0.f;
#pragma unroll 16
    for (int d = 0; d < 64; ++d) e += (float)qp[d] * (float)kp[d];
    e *= 0.03125f;
    // softmax over g: reduction within 16-lane groups (g == lane & 15)
    float mx = e;
#pragma unroll
    for (int o = 1; o < 16; o <<= 1) mx = fmaxf(mx, __shfl_xor(mx, o, 16));
    float ex = __expf(e - mx);
    float sm = ex;
#pragma unroll
    for (int o = 1; o < 16; o <<= 1) sm += __shfl_xor(sm, o, 16);
    att[h * 16 + g] = ex / sm;
  }
  __syncthreads();

  // out[h',d'] = sum_g att[h'][g] * v[g,d'];  4 outputs per thread
  float vals[4];
  float s1 = 0.f, s2 = 0.f;
#pragma unroll
  for (int i = 0; i < 4; ++i) {
    const int j  = tid + 256 * i;
    const int hp = j >> 6, dp = j & 63;
    const float*    ap = att + hp * 16;
    const _Float16* vp = vs + dp;
    float s = 0.f;
#pragma unroll
    for (int gg = 0; gg < 16; ++gg) s += ap[gg] * (float)vp[gg * SP];
    vals[i] = s;
    s1 += s;
    s2 += s * s;
  }
  // block reduction for LayerNorm statistics (population variance)
#pragma unroll
  for (int o = 16; o > 0; o >>= 1) {
    s1 += __shfl_xor(s1, o, 32);
    s2 += __shfl_xor(s2, o, 32);
  }
  const int wid = tid >> 5;
  if ((tid & 31) == 0) { red1[wid] = s1; red2[wid] = s2; }
  __syncthreads();
  if (tid == 0) {
    float t1 = 0.f, t2 = 0.f;
#pragma unroll
    for (int w = 0; w < 8; ++w) { t1 += red1[w]; t2 += red2[w]; }
    const float mean = t1 * (1.f / 1024.f);
    const float var  = t2 * (1.f / 1024.f) - mean * mean;
    stat[0] = mean;
    stat[1] = rsqrtf(var + 1e-5f);
  }
  __syncthreads();
  const float mean = stat[0], rstd = stat[1];
#pragma unroll
  for (int i = 0; i < 4; ++i) {
    const int j = tid + 256 * i;
    const float vn = (vals[i] - mean) * rstd * ln_g[j] + ln_b[j];
    Outh[base + j] = (_Float16)vn;
  }
}

// ---- K3: output projection: LN(out) @ Wo^T + bo -> fp32 ---------------------
// grid = (ROWS/64, DM/64), block = 32
__global__ __launch_bounds__(32) void out_gemm_kernel(
    const _Float16* __restrict__ Xh, const _Float16* __restrict__ Wo,
    const float* __restrict__ bo, float* __restrict__ Out) {
  const int lane = threadIdx.x;
  const int l16  = lane & 15;
  const int hi   = lane >> 4;
  const int rowbase = blockIdx.x * 64;
  const int colbase = blockIdx.y * 64;

  const _Float16* Xrow[4];
  const _Float16* Wrow[4];
#pragma unroll
  for (int i = 0; i < 4; ++i) Xrow[i] = Xh + (size_t)(rowbase + 16 * i + l16) * DM;
#pragma unroll
  for (int j = 0; j < 4; ++j) Wrow[j] = Wo + (size_t)(colbase + 16 * j + l16) * DM;

  v8f zero = {};
  v8f acc[4][4];
#pragma unroll
  for (int i = 0; i < 4; ++i)
#pragma unroll
    for (int j = 0; j < 4; ++j) acc[i][j] = zero;

  for (int kt = 0; kt < DM; kt += 32) {
    const int ka = kt + hi * 8;
    const int kb = kt + hi * 8;
    v16h a[4];
#pragma unroll
    for (int i = 0; i < 4; ++i) a[i] = a_frag_f16(Xrow[i], ka);
#pragma unroll
    for (int j = 0; j < 4; ++j) {
      v16h b = b_frag_f16(Wrow[j], kb);
#pragma unroll
      for (int i = 0; i < 4; ++i)
        acc[i][j] = __builtin_amdgcn_wmma_f32_16x16x32_f16(
            false, a[i], false, b, (short)0, acc[i][j], false, false);
    }
  }

#pragma unroll
  for (int j = 0; j < 4; ++j) {
    const int col = colbase + 16 * j + l16;
    const float bias = bo[col];
#pragma unroll
    for (int i = 0; i < 4; ++i) {
      float* orow = Out + (size_t)(rowbase + 16 * i + hi * 8) * DM + col;
#pragma unroll
      for (int r = 0; r < 8; ++r)
        orow[(size_t)r * DM] = acc[i][j][r] + bias;
    }
  }
}

// ---- launcher ---------------------------------------------------------------
extern "C" void kernel_launch(void* const* d_in, const int* in_sizes, int n_in,
                              void* d_out, int out_size, void* d_ws, size_t ws_size,
                              hipStream_t stream) {
  (void)in_sizes; (void)n_in; (void)out_size; (void)ws_size;

  const float* query = (const float*)d_in[0];
  const float* key   = (const float*)d_in[1];
  const float* value = (const float*)d_in[2];
  const float* Wq    = (const float*)d_in[3];
  const float* bq    = (const float*)d_in[4];
  const float* Wk    = (const float*)d_in[5];
  const float* bk    = (const float*)d_in[6];
  const float* Wv    = (const float*)d_in[7];
  const float* bv    = (const float*)d_in[8];
  const float* ln_g  = (const float*)d_in[9];
  const float* ln_b  = (const float*)d_in[10];
  const float* Wo    = (const float*)d_in[11];
  const float* bo    = (const float*)d_in[12];
  float* out = (float*)d_out;

  const size_t elems  = (size_t)ROWS * DM;  // 16.7M halves per activation buffer
  const size_t welems = (size_t)DM * DM;    // 1M halves per weight matrix
  _Float16* qh  = (_Float16*)d_ws;
  _Float16* kh  = qh + elems;
  _Float16* vh  = kh + elems;
  _Float16* lnh = vh + elems;
  _Float16* whq = lnh + elems;
  _Float16* whk = whq + welems;
  _Float16* whv = whk + welems;
  _Float16* who = whv + welems;   // total ws use: 136 MB

  // K0: weights fp32 -> f16 (one pass, 4 matrices)
  dim3 g0((unsigned)(welems / (256 * 8)), 4, 1);
  cvt_w_kernel<<<g0, 256, 0, stream>>>(Wq, Wk, Wv, Wo, whq, whk, whv, who);

  dim3 g1(ROWS / 64, DM / 64, 3);
  qkv_gemm_kernel<<<g1, 32, 0, stream>>>(query, key, value, whq, whk, whv,
                                         bq, bk, bv, qh, kh, vh);

  attn_ln_kernel<<<ROWS, 256, 0, stream>>>(qh, kh, vh, ln_g, ln_b, lnh);

  dim3 g3(ROWS / 64, DM / 64, 1);
  out_gemm_kernel<<<g3, 32, 0, stream>>>(lnh, who, bo, out);
}
